// RANDIPShiftedLoss_5231270167087
// MI455X (gfx1250) — matrix operand order repaired and checked by reading, not compiled, not run on hardware
//
#include <hip/hip_runtime.h>
#include <hip/hip_bf16.h>
#include <math.h>

typedef __attribute__((ext_vector_type(16))) _Float16 v16h;
typedef __attribute__((ext_vector_type(8)))  _Float16 v8h;
typedef __attribute__((ext_vector_type(8)))  float    v8f;

#define NANG    9      // angles -4..+4 deg
#define SS      16     // subtile size
#define NP      256    // pixels per subtile
#define NSHIFT  81     // (2*4+1)^2 shifts
#define NSROW   96     // 81 padded to 6 WMMA row-tiles
#define ASTRIDE 264    // LDS row stride in halves (16B aligned, 4-bank skew)
#define FLT_MAX_F 3.402823466e+38f

// ---------------------------------------------------------------------------
// Setup: per angle build the 256x256 fp16 bilinear-rotation operator R_a
// (row-major: R[p][q] = weight of source pixel q for rotated output pixel p),
// the mask m_a[p] (bilinear of ones), and 1/sum(m_a).
// ---------------------------------------------------------------------------
__global__ void rot_setup_kernel(_Float16* __restrict__ R,
                                 float* __restrict__ m,
                                 float* __restrict__ inv_msum) {
  int a = blockIdx.x;            // 0..8 -> angle (a-4) degrees
  int p = threadIdx.x;           // 0..255 output pixel (i = p>>4, j = p&15)
  _Float16* Ra = R + (size_t)a * NP * NP;
  for (int idx = p; idx < NP * NP; idx += 256) Ra[idx] = (_Float16)0.0f;
  __syncthreads();

  int i = p >> 4, j = p & 15;
  float t  = (float)(a - 4) * 0.017453292519943295f;  // deg2rad
  float ca = cosf(t), sa = sinf(t);
  const float cx = 8.0f, cy = 8.0f;                   // ss/2
  float sx = ca * (float)j - sa * (float)i + (1.0f - ca) * cx + sa * cy;
  float sy = sa * (float)j + ca * (float)i - sa * cx + (1.0f - ca) * cy;
  float x0f = floorf(sx), y0f = floorf(sy);
  float wx1 = sx - x0f, wy1 = sy - y0f;
  float wx0 = 1.0f - wx1, wy0 = 1.0f - wy1;
  int x0 = (int)x0f, y0 = (int)y0f;

  _Float16* row = Ra + (size_t)p * NP;   // thread owns this row exclusively
  float mval = 0.0f;
  float w4[4] = {wy0 * wx0, wy0 * wx1, wy1 * wx0, wy1 * wx1};
  int   x4[4] = {x0, x0 + 1, x0,     x0 + 1};
  int   y4[4] = {y0, y0,     y0 + 1, y0 + 1};
#pragma unroll
  for (int k = 0; k < 4; ++k) {
    int xx = x4[k], yy = y4[k];
    if (xx >= 0 && xx < SS && yy >= 0 && yy < SS) {
      int q = yy * 16 + xx;
      row[q] = (_Float16)((float)row[q] + w4[k]);
      mval += w4[k];
    }
  }
  m[a * NP + p] = mval;

  __shared__ float red[256];
  red[p] = mval;
  __syncthreads();
  for (int s = 128; s > 0; s >>= 1) {
    if (p < s) red[p] += red[p + s];
    __syncthreads();
  }
  if (p == 0) inv_msum[a] = 1.0f / red[0];
}

// ---------------------------------------------------------------------------
// Main: one block per (batch, subtile). GEMM  RB[96x256] = A[96x256] * R_a^T
// via v_wmma_f32_16x16x32_f16, then masked-SSE epilogue + min over shifts.
//
// Wave tiling: each of 8 waves owns ALL 6 mt row-tiles x 2 nt column-tiles
// (12 accumulator tiles = 96 VGPRs). Each B fragment (global, L2-resident R)
// feeds 6 WMMAs in registers -> one read of R_a per block per angle (128 KB),
// ~6x less L2 traffic than a 1-mt-per-wave mapping; A comes from LDS.
// ---------------------------------------------------------------------------
__global__ void __launch_bounds__(256)
randip_main_kernel(const float* __restrict__ fm1,
                   const float* __restrict__ fm2,
                   const _Float16* __restrict__ R,
                   const float* __restrict__ m,
                   const float* __restrict__ inv_msum,
                   float* __restrict__ partials) {
  __shared__ _Float16 ldsA[NSROW * ASTRIDE];  // shifted fm2 patches, fp16
  __shared__ float lds_r1[NP];                // fm1 subtile, f32
  __shared__ float lds_part[8][NSROW];        // per-wave per-shift partial SSE
  __shared__ float lds_minval[NSROW];         // running min over angles

  int tid  = threadIdx.x;
  int bid  = blockIdx.x;
  int b    = bid >> 4;
  int st   = bid & 15;
  int tx   = (st & 3) * SS;
  int ty   = (st >> 2) * SS;
  int wave = tid >> 5;
  int lane = tid & 31;
  int r    = lane & 15;   // row-within-tile (A) / column-within-tile (B,C)
  int hi   = lane >> 4;   // half-wave selector

  const float* f1 = fm1 + (size_t)b * 4096;
  const float* f2 = fm2 + (size_t)b * 4096;

  lds_r1[tid] = f1[(ty + (tid >> 4)) * 64 + tx + (tid & 15)];
  if (tid < NSROW) lds_minval[tid] = FLT_MAX_F;

  // Build A: row s = shifted patch (dw = s/9-4 outer, dh = s%9-4 inner),
  // column q = pixel (q>>4, q&15). Rows 81..95 zero-padded.
  for (int idx = tid; idx < NSROW * NP; idx += 256) {
    int s = idx >> 8, q = idx & 255;
    float v = 0.0f;
    if (s < NSHIFT) {
      int dw = s / 9 - 4, dh = s % 9 - 4;
      int y0 = min(max(ty + dh, 0), 48);
      int x0 = min(max(tx + dw, 0), 48);
      v = f2[(y0 + (q >> 4)) * 64 + x0 + (q & 15)];
    }
    ldsA[s * ASTRIDE + q] = (_Float16)v;
  }
  __syncthreads();

  int nt0 = wave * 2;              // this wave's two output-pixel column tiles
  int p0  = (nt0 + 0) * 16 + r;    // lane's columns within those tiles
  int p1  = (nt0 + 1) * 16 + r;

  for (int a = 0; a < NANG; ++a) {
    const _Float16* Ra = R + (size_t)a * NP * NP;
    const float*    ma = m + a * NP;
    const _Float16* B0 = Ra + (size_t)p0 * NP + hi * 16;  // B = R^T columns
    const _Float16* B1 = Ra + (size_t)p1 * NP + hi * 16;

    const v8f vzero = {0.f, 0.f, 0.f, 0.f, 0.f, 0.f, 0.f, 0.f};
    v8f acc[6][2];
#pragma unroll
    for (int mt = 0; mt < 6; ++mt) {
      acc[mt][0] = vzero;
      acc[mt][1] = vzero;
    }

#pragma unroll
    for (int kk = 0; kk < 8; ++kk) {
      // B frag (32x16 f16): lane = column p, K = hi*16 .. hi*16+15 contiguous
      // along R's row p. Loaded once, reused by all 6 mt tiles.
      v16h bf0 = *(const v16h*)(B0 + kk * 32);
      v16h bf1 = *(const v16h*)(B1 + kk * 32);
#pragma unroll
      for (int mt = 0; mt < 6; ++mt) {
        // A frag (16x32 f16): lane holds row r; hi=0 -> K {0..15},
        // hi=1 -> K {16..31} of this 32-wide K step.
        const _Float16* Arow = &ldsA[(mt * 16 + r) * ASTRIDE];
        v8h alo = *(const v8h*)(Arow + kk * 32 + hi * 16);
        v8h ahi = *(const v8h*)(Arow + kk * 32 + hi * 16 + 8);
        v16h af = __builtin_shufflevector(alo, ahi,
            0, 1, 2, 3, 4, 5, 6, 7, 8, 9, 10, 11, 12, 13, 14, 15);
        acc[mt][0] = __builtin_amdgcn_wmma_f32_16x16x32_f16(
            false, af, false, bf0, (short)0, acc[mt][0], false, false);
        acc[mt][1] = __builtin_amdgcn_wmma_f32_16x16x32_f16(
            false, af, false, bf1, (short)0, acc[mt][1], false, false);
      }
    }

    // Epilogue: C layout -> VGPR v holds row (hi*8+v), lane holds column p.
    float m0  = ma[p0],     m1  = ma[p1];
    float r10 = lds_r1[p0], r11 = lds_r1[p1];
#pragma unroll
    for (int mt = 0; mt < 6; ++mt) {
#pragma unroll
      for (int v = 0; v < 8; ++v) {
        float d0 = r10 - acc[mt][0][v];
        float d1 = r11 - acc[mt][1][v];
        float val = m0 * d0 * d0 + m1 * d1 * d1;
        for (int off = 1; off < 16; off <<= 1)
          val += __shfl_xor(val, off, 32);
        if (r == 0) lds_part[wave][mt * 16 + hi * 8 + v] = val;
      }
    }
    __syncthreads();
    if (tid < NSROW) {
      float sum = 0.0f;
#pragma unroll
      for (int w = 0; w < 8; ++w) sum += lds_part[w][tid];  // fixed order
      float val = (tid < NSHIFT) ? sum * inv_msum[a] : FLT_MAX_F;
      lds_minval[tid] = fminf(lds_minval[tid], val);
    }
    __syncthreads();
  }

  if (tid == 0) {
    float mn = FLT_MAX_F;
    for (int s = 0; s < NSHIFT; ++s) mn = fminf(mn, lds_minval[s]);
    partials[(size_t)b * 16 + st] = mn;
  }
}

// ---------------------------------------------------------------------------
// Finalize: deterministic fixed-order sum of 16 subtile partials per batch.
// ---------------------------------------------------------------------------
__global__ void randip_finalize_kernel(const float* __restrict__ partials,
                                       float* __restrict__ out, int bs) {
  int b = blockIdx.x * 256 + threadIdx.x;
  if (b < bs) {
    float s = 0.0f;
#pragma unroll
    for (int i = 0; i < 16; ++i) s += partials[(size_t)b * 16 + i];
    out[b] = s;
  }
}

// ---------------------------------------------------------------------------
extern "C" void kernel_launch(void* const* d_in, const int* in_sizes, int n_in,
                              void* d_out, int out_size, void* d_ws, size_t ws_size,
                              hipStream_t stream) {
  (void)n_in; (void)out_size; (void)ws_size;
  const float* fm1 = (const float*)d_in[0];
  const float* fm2 = (const float*)d_in[1];
  int bs = in_sizes[0] / 4096;                 // (bs,1,64,64)

  char* ws = (char*)d_ws;
  // Workspace layout (bytes):
  //   R:        9 * 256*256 * 2 = 1,179,648
  //   m:        9 * 256 * 4     =     9,216   @ 1,179,648
  //   inv_msum: 9 * 4 (padded)             @ 1,188,864
  //   partials: bs * 16 * 4                @ 1,188,928
  _Float16* R        = (_Float16*)ws;
  float*    msk      = (float*)(ws + 1179648);
  float*    inv_msum = (float*)(ws + 1188864);
  float*    partials = (float*)(ws + 1188928);
  float*    out      = (float*)d_out;

  rot_setup_kernel<<<NANG, 256, 0, stream>>>(R, msk, inv_msum);
  randip_main_kernel<<<bs * 16, 256, 0, stream>>>(fm1, fm2, R, msk, inv_msum,
                                                  partials);
  randip_finalize_kernel<<<(bs + 255) / 256, 256, 0, stream>>>(partials, out, bs);
}
